// GNNEventDetector_2894807957940
// MI455X (gfx1250) — compile-verified, matrix operand order checked by reading, not run on hardware
//
#include <hip/hip_runtime.h>
#include <hip/hip_bf16.h>

#define N_NODES 100000
#define N_EDGES 1600000
#define E_TOT   (N_EDGES + N_NODES)   // self-loops appended
#define N_GRAPHS 64

typedef __attribute__((ext_vector_type(16))) _Float16 v16h;
typedef __attribute__((ext_vector_type(8)))  float    v8f;

// ---------------- helpers ----------------

__device__ __forceinline__ float lrelu(float v) { return v > 0.f ? v : 0.2f * v; }
__device__ __forceinline__ float elu(float v)   { return v > 0.f ? v : (__expf(v) - 1.f); }

__device__ __forceinline__ void atomicMaxF(float* addr, float v) {
    unsigned int* ua = (unsigned int*)addr;
    unsigned int old = *ua;
    while (__uint_as_float(old) < v) {
        unsigned int assumed = old;
        old = atomicCAS(ua, assumed, __float_as_uint(v));
        if (old == assumed) break;
    }
}

// K index for element i (0..15) of the f16 A/B operand per CDNA5 ISA layout
// (16-bit A 16x32: VGPR0..3 hold K {0..7} (+8 for lanes 16..31), VGPR4..7 hold K {16..23}(+8))
__device__ __forceinline__ int wmma_kidx(int i, int half) {
    int p = i >> 1, lo = i & 1;
    int kb = (p < 4) ? (2 * p) : (16 + 2 * (p - 4));
    return kb + half * 8 + lo;
}

// ---------------- generic fill ----------------

__global__ void fill_kernel(float* __restrict__ p, long long n, float v) {
    long long i = (long long)blockIdx.x * blockDim.x + threadIdx.x;
    if (i < n) p[i] = v;
}

// ---------------- GEMM1: h1[N,128] = x[N,32] @ W1[32,128]  (f16 out, WMMA) ----------------

__global__ __launch_bounds__(32)
void gemm1_kernel(const float* __restrict__ x, const float* __restrict__ W1,
                  _Float16* __restrict__ h1) {
    const int nodeBase = blockIdx.x * 16;
    const int lane = threadIdx.x;
    const int m    = lane & 15;
    const int half = lane >> 4;

    v16h a;
    #pragma unroll
    for (int i = 0; i < 16; ++i) {
        int k = wmma_kidx(i, half);
        a[i] = (_Float16)x[(nodeBase + m) * 32 + k];
    }
    #pragma unroll
    for (int t = 0; t < 8; ++t) {
        const int n0 = t * 16;
        v16h b;
        #pragma unroll
        for (int i = 0; i < 16; ++i) {
            int k = wmma_kidx(i, half);
            b[i] = (_Float16)W1[k * 128 + n0 + m];
        }
        v8f c = {};
        c = __builtin_amdgcn_wmma_f32_16x16x32_f16(false, a, false, b, (short)0, c, false, false);
        #pragma unroll
        for (int r = 0; r < 8; ++r) {
            int M = r + half * 8;
            h1[(size_t)(nodeBase + M) * 128 + n0 + m] = (_Float16)c[r];
        }
    }
}

// ---------------- GEMM2: h2[N,64] = h2in[N,128] @ W2[128,64]  (f16 in/out, WMMA) ----------------

__global__ __launch_bounds__(32)
void gemm2_kernel(const _Float16* __restrict__ h2in, const float* __restrict__ W2,
                  _Float16* __restrict__ h2) {
    const int nodeBase = blockIdx.x * 16;
    const int lane = threadIdx.x;
    const int m    = lane & 15;
    const int half = lane >> 4;

    v8f acc[4];
    #pragma unroll
    for (int t = 0; t < 4; ++t) acc[t] = (v8f){};

    #pragma unroll
    for (int kk = 0; kk < 4; ++kk) {
        const int kbase = kk * 32;
        v16h a;
        #pragma unroll
        for (int i = 0; i < 16; ++i) {
            int k = kbase + wmma_kidx(i, half);
            a[i] = h2in[(size_t)(nodeBase + m) * 128 + k];
        }
        #pragma unroll
        for (int t = 0; t < 4; ++t) {
            const int n0 = t * 16;
            v16h b;
            #pragma unroll
            for (int i = 0; i < 16; ++i) {
                int k = kbase + wmma_kidx(i, half);
                b[i] = (_Float16)W2[k * 64 + n0 + m];
            }
            acc[t] = __builtin_amdgcn_wmma_f32_16x16x32_f16(false, a, false, b, (short)0,
                                                            acc[t], false, false);
        }
    }
    #pragma unroll
    for (int t = 0; t < 4; ++t) {
        #pragma unroll
        for (int r = 0; r < 8; ++r) {
            int M = r + half * 8;
            h2[(size_t)(nodeBase + M) * 64 + t * 16 + m] = (_Float16)acc[t][r];
        }
    }
}

// ---------------- attention dot products ----------------

__global__ void att1_kernel(const _Float16* __restrict__ h1,
                            const float* __restrict__ att_src, const float* __restrict__ att_dst,
                            float* __restrict__ as1, float* __restrict__ ad1) {
    int tid = blockIdx.x * blockDim.x + threadIdx.x;
    if (tid >= N_NODES * 2) return;
    int n = tid >> 1, hh = tid & 1;
    const _Float16* hp = h1 + (size_t)n * 128 + hh * 64;
    float s = 0.f, d = 0.f;
    #pragma unroll 8
    for (int c = 0; c < 64; ++c) {
        float hv = (float)hp[c];
        s += hv * att_src[hh * 64 + c];
        d += hv * att_dst[hh * 64 + c];
    }
    as1[tid] = s; ad1[tid] = d;
}

__global__ void att2_kernel(const _Float16* __restrict__ h2,
                            const float* __restrict__ att_src, const float* __restrict__ att_dst,
                            float* __restrict__ as2, float* __restrict__ ad2) {
    int n = blockIdx.x * blockDim.x + threadIdx.x;
    if (n >= N_NODES) return;
    const _Float16* hp = h2 + (size_t)n * 64;
    float s = 0.f, d = 0.f;
    #pragma unroll 8
    for (int c = 0; c < 64; ++c) {
        float hv = (float)hp[c];
        s += hv * att_src[c];
        d += hv * att_dst[c];
    }
    as2[n] = s; ad2[n] = d;
}

// ---------------- edge pass 1: segment max of attention logits ----------------

__global__ void edge_max1(const long long* __restrict__ ei,
                          const float* __restrict__ as1, const float* __restrict__ ad1,
                          float* __restrict__ m1) {
    long long tid = (long long)blockIdx.x * blockDim.x + threadIdx.x;
    if (tid >= (long long)E_TOT * 2) return;
    int e = (int)(tid >> 1), hh = (int)(tid & 1);
    int s, d;
    if (e < N_EDGES) { s = (int)ei[e]; d = (int)ei[N_EDGES + e]; }
    else             { s = d = e - N_EDGES; }
    float v = lrelu(as1[s * 2 + hh] + ad1[d * 2 + hh]);
    atomicMaxF(&m1[d * 2 + hh], v);
}

__global__ void edge_max2(const long long* __restrict__ ei,
                          const float* __restrict__ as2, const float* __restrict__ ad2,
                          float* __restrict__ m2) {
    long long tid = (long long)blockIdx.x * blockDim.x + threadIdx.x;
    if (tid >= (long long)E_TOT) return;
    int e = (int)tid;
    int s, d;
    if (e < N_EDGES) { s = (int)ei[e]; d = (int)ei[N_EDGES + e]; }
    else             { s = d = e - N_EDGES; }
    float v = lrelu(as2[s] + ad2[d]);
    atomicMaxF(&m2[d], v);
}

// ---------------- edge pass 2: accumulate denom + unnormalized messages ----------------
// 8 threads per edge (head = sub>>2, 16-channel chunk = sub&3)

__global__ void edge_agg1(const long long* __restrict__ ei,
                          const float* __restrict__ as1, const float* __restrict__ ad1,
                          const float* __restrict__ m1, float* __restrict__ den1,
                          const _Float16* __restrict__ h1, float* __restrict__ out1) {
    long long tid = (long long)blockIdx.x * blockDim.x + threadIdx.x;
    if (tid >= (long long)E_TOT * 8) return;
    int e   = (int)(tid >> 3);
    int sub = (int)(tid & 7);
    int hh  = sub >> 2;
    int q   = sub & 3;
    int s, d;
    if (e < N_EDGES) { s = (int)ei[e]; d = (int)ei[N_EDGES + e]; }
    else             { s = d = e - N_EDGES; }
    float v = lrelu(as1[s * 2 + hh] + ad1[d * 2 + hh]);
    float w = __expf(v - m1[d * 2 + hh]);
    if (q == 0) atomicAdd(&den1[d * 2 + hh], w);
    const _Float16* hp = h1   + (size_t)s * 128 + hh * 64 + q * 16;
    float*          op = out1 + (size_t)d * 128 + hh * 64 + q * 16;
    #pragma unroll
    for (int c = 0; c < 16; ++c)
        atomicAdd(op + c, w * (float)hp[c]);
}

// 4 threads per edge (single head, 16-channel chunks)
__global__ void edge_agg2(const long long* __restrict__ ei,
                          const float* __restrict__ as2, const float* __restrict__ ad2,
                          const float* __restrict__ m2, float* __restrict__ den2,
                          const _Float16* __restrict__ h2, float* __restrict__ out2) {
    long long tid = (long long)blockIdx.x * blockDim.x + threadIdx.x;
    if (tid >= (long long)E_TOT * 4) return;
    int e = (int)(tid >> 2);
    int q = (int)(tid & 3);
    int s, d;
    if (e < N_EDGES) { s = (int)ei[e]; d = (int)ei[N_EDGES + e]; }
    else             { s = d = e - N_EDGES; }
    float v = lrelu(as2[s] + ad2[d]);
    float w = __expf(v - m2[d]);
    if (q == 0) atomicAdd(&den2[d], w);
    const _Float16* hp = h2   + (size_t)s * 64 + q * 16;
    float*          op = out2 + (size_t)d * 64 + q * 16;
    #pragma unroll
    for (int c = 0; c < 16; ++c)
        atomicAdd(op + c, w * (float)hp[c]);
}

// ---------------- layer-1 finalize: normalize + bias + ELU -> f16 features for layer 2 ----------------

__global__ void fin1_kernel(const float* __restrict__ out1, const float* __restrict__ den1,
                            const float* __restrict__ b1, _Float16* __restrict__ h2in) {
    long long tid = (long long)blockIdx.x * blockDim.x + threadIdx.x;
    if (tid >= (long long)N_NODES * 128) return;
    int n = (int)(tid >> 7), f = (int)(tid & 127);
    float v = out1[tid] / (den1[n * 2 + (f >> 6)] + 1e-16f) + b1[f];
    h2in[tid] = (_Float16)elu(v);
}

// ---------------- layer-2 finalize fused with per-graph max/sum pooling ----------------

__global__ void pool_kernel(const float* __restrict__ out2, const float* __restrict__ den2,
                            const float* __restrict__ b2, const long long* __restrict__ batch,
                            float* __restrict__ gmax, float* __restrict__ gsum,
                            float* __restrict__ gcnt) {
    long long tid = (long long)blockIdx.x * blockDim.x + threadIdx.x;
    if (tid >= (long long)N_NODES * 64) return;
    int n = (int)(tid >> 6), c = (int)(tid & 63);
    float v = elu(out2[tid] / (den2[n] + 1e-16f) + b2[c]);
    int g = (int)batch[n];
    atomicMaxF(&gmax[g * 64 + c], v);
    atomicAdd(&gsum[g * 64 + c], v);
    if (c == 0) atomicAdd(&gcnt[g], 1.f);
}

// ---------------- per-graph MLP head: relu(g@lin1+b) @ lin2 -> sigmoid ----------------

__global__ __launch_bounds__(64)
void head_kernel(const float* __restrict__ gmax, const float* __restrict__ gsum,
                 const float* __restrict__ gcnt,
                 const float* __restrict__ lin1W, const float* __restrict__ lin1b,
                 const float* __restrict__ lin2W, const float* __restrict__ lin2b,
                 float* __restrict__ out) {
    int g = threadIdx.x;
    if (g >= N_GRAPHS) return;
    float gv[64];
    float cnt = fmaxf(gcnt[g], 1.0f);
    #pragma unroll 8
    for (int c = 0; c < 64; ++c)
        gv[c] = gmax[g * 64 + c] + gsum[g * 64 + c] / cnt;
    float acc = lin2b[0];
    for (int j = 0; j < 64; ++j) {
        float s = lin1b[j];
        #pragma unroll 8
        for (int c = 0; c < 64; ++c)
            s += gv[c] * lin1W[c * 64 + j];
        s = fmaxf(s, 0.f);
        acc += s * lin2W[j];
    }
    out[g] = 1.f / (1.f + __expf(-acc));
}

// ---------------- launcher ----------------

static inline int gblocks(long long n, int bs) { return (int)((n + bs - 1) / bs); }

extern "C" void kernel_launch(void* const* d_in, const int* in_sizes, int n_in,
                              void* d_out, int out_size, void* d_ws, size_t ws_size,
                              hipStream_t stream) {
    const float*     x        = (const float*)d_in[0];
    const long long* ei       = (const long long*)d_in[1];
    const long long* batch    = (const long long*)d_in[2];
    const float*     W1       = (const float*)d_in[3];
    const float*     att_src1 = (const float*)d_in[4];
    const float*     att_dst1 = (const float*)d_in[5];
    const float*     b1       = (const float*)d_in[6];
    const float*     W2       = (const float*)d_in[7];
    const float*     att_src2 = (const float*)d_in[8];
    const float*     att_dst2 = (const float*)d_in[9];
    const float*     b2       = (const float*)d_in[10];
    const float*     lin1W    = (const float*)d_in[11];
    const float*     lin1b    = (const float*)d_in[12];
    const float*     lin2W    = (const float*)d_in[13];
    const float*     lin2b    = (const float*)d_in[14];
    float* out = (float*)d_out;

    char* ws = (char*)d_ws;
    // workspace carve (all offsets 128B-aligned)
    _Float16* h1   = (_Float16*)(ws + 0);                     // N*128 f16  (reused as h2in)
    float*    out1 = (float*)   (ws + 25600000);              // N*128 f32
    float*    as1  = (float*)   (ws + 76800000);              // N*2
    float*    ad1  = (float*)   (ws + 77600000);              // N*2
    float*    m1   = (float*)   (ws + 78400000);              // N*2
    float*    den1 = (float*)   (ws + 79200000);              // N*2
    _Float16* h2   = (_Float16*)(ws + 80000000);              // N*64 f16
    float*    out2 = (float*)   (ws + 92800000);              // N*64 f32
    float*    as2  = (float*)   (ws + 118400000);             // N
    float*    ad2  = (float*)   (ws + 118800000);             // N
    float*    m2   = (float*)   (ws + 119200000);             // N
    float*    den2 = (float*)   (ws + 119600000);             // N
    float*    gmax = (float*)   (ws + 120000000);             // 64*64
    float*    gsum = (float*)   (ws + 120016384);             // 64*64
    float*    gcnt = (float*)   (ws + 120032768);             // 64
    _Float16* h2in = h1;                                      // layer-2 input reuses h1 storage

    const int BS = 256;
    const float NEG_INF = -__builtin_inff();

    // init accumulators
    fill_kernel<<<gblocks((long long)N_NODES * 2, BS), BS, 0, stream>>>(m1,  (long long)N_NODES * 2, NEG_INF);
    fill_kernel<<<gblocks((long long)N_NODES * 2, BS), BS, 0, stream>>>(den1,(long long)N_NODES * 2, 0.f);
    fill_kernel<<<gblocks((long long)N_NODES * 128, BS), BS, 0, stream>>>(out1,(long long)N_NODES * 128, 0.f);
    fill_kernel<<<gblocks((long long)N_NODES, BS), BS, 0, stream>>>(m2,  (long long)N_NODES, NEG_INF);
    fill_kernel<<<gblocks((long long)N_NODES, BS), BS, 0, stream>>>(den2,(long long)N_NODES, 0.f);
    fill_kernel<<<gblocks((long long)N_NODES * 64, BS), BS, 0, stream>>>(out2,(long long)N_NODES * 64, 0.f);
    fill_kernel<<<gblocks(64 * 64, BS), BS, 0, stream>>>(gmax, 64 * 64, NEG_INF);
    fill_kernel<<<gblocks(64 * 64, BS), BS, 0, stream>>>(gsum, 64 * 64, 0.f);
    fill_kernel<<<1, 64, 0, stream>>>(gcnt, 64, 0.f);

    // ---- layer 1 ----
    gemm1_kernel<<<N_NODES / 16, 32, 0, stream>>>(x, W1, h1);
    att1_kernel<<<gblocks((long long)N_NODES * 2, BS), BS, 0, stream>>>(h1, att_src1, att_dst1, as1, ad1);
    edge_max1<<<gblocks((long long)E_TOT * 2, BS), BS, 0, stream>>>(ei, as1, ad1, m1);
    edge_agg1<<<gblocks((long long)E_TOT * 8, BS), BS, 0, stream>>>(ei, as1, ad1, m1, den1, h1, out1);
    fin1_kernel<<<gblocks((long long)N_NODES * 128, BS), BS, 0, stream>>>(out1, den1, b1, h2in);

    // ---- layer 2 ----
    gemm2_kernel<<<N_NODES / 16, 32, 0, stream>>>(h2in, W2, h2);
    att2_kernel<<<gblocks((long long)N_NODES, BS), BS, 0, stream>>>(h2, att_src2, att_dst2, as2, ad2);
    edge_max2<<<gblocks((long long)E_TOT, BS), BS, 0, stream>>>(ei, as2, ad2, m2);
    edge_agg2<<<gblocks((long long)E_TOT * 4, BS), BS, 0, stream>>>(ei, as2, ad2, m2, den2, h2, out2);

    // ---- pooling + MLP head ----
    pool_kernel<<<gblocks((long long)N_NODES * 64, BS), BS, 0, stream>>>(out2, den2, b2, batch, gmax, gsum, gcnt);
    head_kernel<<<1, 64, 0, stream>>>(gmax, gsum, gcnt, lin1W, lin1b, lin2W, lin2b, out);
}
